// Inv2d_44135083934186
// MI455X (gfx1250) — matrix operand order repaired and checked by reading, not compiled, not run on hardware
//
#include <hip/hip_runtime.h>
#include <hip/hip_bf16.h>

typedef __attribute__((ext_vector_type(16))) _Float16 v16h;
typedef __attribute__((ext_vector_type(8)))  _Float16 v8h;
typedef __attribute__((ext_vector_type(8)))  float    v8f;

#define HW   3136      // 56*56
#define CIN  256
#define CR   64
#define NK   144       // G*K*K = 16*9
#define LDR  72        // LDS row stride in halves (144 bytes, 16B-aligned steps)

// ---------------------------------------------------------------------------
// Kernel 0: convert weights fp32 -> fp16 into workspace
// ---------------------------------------------------------------------------
__global__ void cvt_weights(const float* __restrict__ rw,
                            const float* __restrict__ sw,
                            _Float16* __restrict__ rwh,
                            _Float16* __restrict__ swh) {
    int i = blockIdx.x * 256 + threadIdx.x;
    if (i < CR * CIN) rwh[i] = (_Float16)rw[i];
    if (i < NK * CR)  swh[i] = (_Float16)sw[i];
}

// ---------------------------------------------------------------------------
// Kernel 1: kernel generation.  ker[p][n] = span_w @ (reduce_w @ x[p]) + b
// One wave == one 16-pixel tile (M=16).  8 waves / workgroup.
// GEMM1: M16 x N64 x K256 (4 n-tiles, 8 k-steps of wmma 16x16x32 f16)
// GEMM2: M16 x N144 x K64  (9 n-tiles, 2 k-steps), r staged through LDS as f16
// ---------------------------------------------------------------------------
__global__ __launch_bounds__(256) void kergen(
    const float*   __restrict__ x,
    const _Float16* __restrict__ rwh,
    const _Float16* __restrict__ swh,
    const float*   __restrict__ span_b,
    float*         __restrict__ ker)
{
    __shared__ _Float16 lds[8][16 * LDR];   // wave-private r staging (18 KB)

    const int wave = threadIdx.x >> 5;
    const int lane = threadIdx.x & 31;
    const int mlo  = lane & 15;     // row / col index within 16
    const int hi   = lane >> 4;     // half-wave selector

    const int tile = blockIdx.x * 8 + wave;     // 3136 tiles total
    const int p0   = tile * 16;                 // first pixel of tile
    const int b    = p0 / HW;
    const int s0   = p0 % HW;                   // never crosses batch (3136%16==0)

    // ---------------- GEMM1 ----------------
    v8f acc1[4] = {};
    const float* xa = x + ((size_t)b * CIN) * HW + s0 + mlo;   // + c*HW per channel

    for (int ks = 0; ks < 8; ++ks) {
        const int k0 = ks * 32;
        // A fragment: lane holds row M=mlo; lanes<16 get K {0..7,16..23},
        // lanes>=16 get K {8..15,24..31} (relative to k0).
        v16h a;
        const int cbase = k0 + hi * 8;
#pragma unroll
        for (int e = 0; e < 16; ++e) {
            int c = cbase + (e < 8 ? e : e + 8);
            a[e] = (_Float16)xa[(size_t)c * HW];
        }
#pragma unroll
        for (int nt = 0; nt < 4; ++nt) {
            // B fragment: lane -> N = nt*16+mlo, K {0..7,16..23} (+8 for hi half)
            const _Float16* bp = rwh + (nt * 16 + mlo) * CIN + k0 + hi * 8;
            v8h blo = *(const v8h*)bp;
            v8h bhi = *(const v8h*)(bp + 16);
            v16h bf = __builtin_shufflevector(blo, bhi,
                        0,1,2,3,4,5,6,7,8,9,10,11,12,13,14,15);
            acc1[nt] = __builtin_amdgcn_wmma_f32_16x16x32_f16(
                false, a, false, bf, (short)0, acc1[nt], false, false);
        }
    }

    // Spill r (16x64) to LDS as fp16 in A-fragment-friendly [m][k] layout.
    // C layout: VGPR v, lanes 0-15 -> (M=v, N=mlo); lanes 16-31 -> (M=8+v, N=mlo)
    _Float16* lrow = &lds[wave][0];
#pragma unroll
    for (int nt = 0; nt < 4; ++nt) {
#pragma unroll
        for (int v = 0; v < 8; ++v) {
            int mm = v + 8 * hi;
            int kk = nt * 16 + mlo;
            lrow[mm * LDR + kk] = (_Float16)acc1[nt][v];
        }
    }

    // ---------------- GEMM2 ----------------
    v8f acc2[9] = {};
#pragma unroll
    for (int ks = 0; ks < 2; ++ks) {
        // A fragment of r from LDS (b128-aligned: row stride 144B)
        const _Float16* ar = lrow + mlo * LDR + ks * 32 + hi * 8;
        v8h alo = *(const v8h*)ar;          // K k0+{0..7}   (or +{8..15})
        v8h ahi = *(const v8h*)(ar + 16);   // K k0+{16..23} (or +{24..31})
        v16h a = __builtin_shufflevector(alo, ahi,
                    0,1,2,3,4,5,6,7,8,9,10,11,12,13,14,15);
#pragma unroll
        for (int nt = 0; nt < 9; ++nt) {
            const _Float16* bp = swh + (nt * 16 + mlo) * CR + ks * 32 + hi * 8;
            v8h blo = *(const v8h*)bp;
            v8h bhi = *(const v8h*)(bp + 16);
            v16h bf = __builtin_shufflevector(blo, bhi,
                        0,1,2,3,4,5,6,7,8,9,10,11,12,13,14,15);
            acc2[nt] = __builtin_amdgcn_wmma_f32_16x16x32_f16(
                false, a, false, bf, (short)0, acc2[nt], false, false);
        }
    }

    // Bias + store ker[p][n].  Per lane, N is constant across the 8 acc VGPRs.
#pragma unroll
    for (int nt = 0; nt < 9; ++nt) {
        int n = nt * 16 + mlo;
        float bias = span_b[n];
        float* orow = ker + (size_t)p0 * NK + n;
#pragma unroll
        for (int v = 0; v < 8; ++v) {
            int mm = v + 8 * hi;
            orow[(size_t)mm * NK] = acc2[nt][v] + bias;
        }
    }
}

// ---------------------------------------------------------------------------
// Kernel 2: involution aggregation (bandwidth-bound).
// block = (b, group, 4-row spatial tile of 224 px), 224 threads.
// out[b, g*16+c, h, w] = sum_kk ker[p][g*9+kk] * x[b, g*16+c, h+di, w+dj]
// ---------------------------------------------------------------------------
__global__ __launch_bounds__(224) void involution(
    const float* __restrict__ x,
    const float* __restrict__ ker,
    float*       __restrict__ out)
{
    const int blk   = blockIdx.x;
    const int stile = blk % 14;
    const int g     = (blk / 14) % 16;
    const int b     = blk / (14 * 16);
    const int t     = threadIdx.x;
    const int s     = stile * 224 + t;
    const int h     = s / 56;
    const int w     = s % 56;

    float kv[9];
    const float* kp = ker + ((size_t)(b * HW + s)) * NK + g * 9;
#pragma unroll
    for (int kk = 0; kk < 9; ++kk) kv[kk] = kp[kk];

    const float* xb = x   + ((size_t)(b * CIN + g * 16)) * HW;
    float*       ob = out + ((size_t)(b * CIN + g * 16)) * HW + s;

    for (int c = 0; c < 16; ++c) {
        const float* xc = xb + (size_t)c * HW;
        float sum = 0.f;
#pragma unroll
        for (int kk = 0; kk < 9; ++kk) {
            int hh = h + kk / 3 - 1;
            int ww = w + kk % 3 - 1;
            float xv = (hh >= 0 && hh < 56 && ww >= 0 && ww < 56)
                         ? xc[hh * 56 + ww] : 0.f;
            sum = fmaf(kv[kk], xv, sum);
        }
        ob[(size_t)c * HW] = sum;
    }
}

// ---------------------------------------------------------------------------
extern "C" void kernel_launch(void* const* d_in, const int* in_sizes, int n_in,
                              void* d_out, int out_size, void* d_ws, size_t ws_size,
                              hipStream_t stream) {
    const float* x        = (const float*)d_in[0];   // [16,256,56,56]
    const float* reduce_w = (const float*)d_in[1];   // [64,256]
    const float* span_w   = (const float*)d_in[2];   // [144,64]
    const float* span_b   = (const float*)d_in[3];   // [144]
    float* out = (float*)d_out;

    // workspace layout
    _Float16* rwh = (_Float16*)d_ws;                           // 32768 B
    _Float16* swh = (_Float16*)((char*)d_ws + 32768);          // 18432 B
    float*    ker = (float*)  ((char*)d_ws + 65536);           // 50176*144*4 B

    cvt_weights<<<64, 256, 0, stream>>>(reduce_w, span_w, rwh, swh);

    // 3136 pixel-tiles, 8 waves (tiles) per workgroup
    kergen<<<392, 256, 0, stream>>>(x, rwh, swh, span_b, ker);

    // 16 batches * 16 groups * 14 spatial tiles
    involution<<<16 * 16 * 14, 224, 0, stream>>>(x, ker, out);
}
